// LuongAttention_4552665334402
// MI455X (gfx1250) — matrix operand order repaired and checked by reading, not compiled, not run on hardware
//
#include <hip/hip_runtime.h>
#include <hip/hip_bf16.h>

// ---------------------------------------------------------------------------
// Luong general attention for MI455X (gfx1250, wave32, WMMA bf16).
//   K1: proj  = enc @ Wa + bias      (bf16 out -> ws)
//   K2: score = dec @ proj^T         (f32 out -> d_out alignment region)
//   K3: softmax rows (in place on alignment region)
//   K4: ctx   = align @ enc          (f32 out -> d_out context region)
// 64x128 block tile, 32x32 wave tile (4 WMMAs/K-step), software-pipelined
// global->reg->LDS staging, packed bf16 LDS stores via inline v_cvt_pk_bf16_f32.
// ---------------------------------------------------------------------------

typedef __attribute__((ext_vector_type(16))) __bf16 v16bf;
typedef __attribute__((ext_vector_type(8)))  float  v8f;

#define BM 64
#define BN 128
#define BK 32

union Frag16 { v16bf v; unsigned u[8]; };

// One VOP3 instruction: pack two f32 into bf16x2 (RNE), per CDNA5 ISA.
__device__ __forceinline__ unsigned pk2(float lo, float hi) {
    unsigned r;
    asm("v_cvt_pk_bf16_f32 %0, %1, %2" : "=v"(r) : "v"(lo), "v"(hi));
    return r;
}

// MODE 0: A f32, B f32 (Wa), +bias, C -> bf16 (proj in ws)
// MODE 1: A f32 (dec), B bf16 rows-are-N (proj), C -> f32 (scores)
// MODE 2: A f32 (align), B f32 (enc, K x N), C -> f32 (context)
template<int MODE>
__global__ __launch_bounds__(256)
void gemm_bf16_wmma(const float*  __restrict__ A,
                    const float*  __restrict__ Bf,
                    const __bf16* __restrict__ Bh,
                    const float*  __restrict__ bias,
                    float*        __restrict__ Cf,
                    __bf16*       __restrict__ Ch,
                    int N, int K, int ldA, int ldB, int ldC,
                    long aBatch, long bBatch, long cBatch)
{
    __shared__ __align__(16) __bf16 As[BM][BK];   // [m][k]
    __shared__ __align__(16) __bf16 Bs[BN][BK];   // [n][k]  (n-major for frag loads)

    const int z    = blockIdx.z;
    const int tid  = threadIdx.x;
    const int wave = tid >> 5;
    const int lane = tid & 31;
    const int half = lane >> 4;     // 0: lanes 0-15, 1: lanes 16-31
    const int l16  = lane & 15;
    const int warpM = wave & 1;     // 2 waves along M (2*32 = 64)
    const int warpN = wave >> 1;    // 4 waves along N (4*32 = 128)

    const int row0 = blockIdx.y * BM;
    const int col0 = blockIdx.x * BN;

    const float*  Ab  = A + (long)z * aBatch;
    const float*  Bfb = (MODE != 1) ? (Bf + (long)z * bBatch) : nullptr;
    const __bf16* Bhb = (MODE == 1) ? (Bh + (long)z * bBatch) : nullptr;

    v8f acc00 = {}, acc01 = {}, acc10 = {}, acc11 = {};

    // ---- staging thread maps ----
    const int a_r = tid >> 2;             // 0..63
    const int a_c = (tid & 3) * 8;        // 0,8,16,24
    const int b_n = (tid & 31) * 4;       // MODE 0/2: 4 cols
    const int b_k = (tid >> 5) * 4;       // MODE 0/2: 4 k rows
    const int h_n = tid >> 1;             // MODE 1: 0..127
    const int h_k = (tid & 1) * 16;       // MODE 1: 16 k elems

    // ---- pipeline registers ----
    float ra[8];
    float rb[4][4];        // MODE 0/2: [k][n]
    uint4 rh[2];           // MODE 1

    auto loadTile = [&](int k0) {
        const float* sA = Ab + (long)(row0 + a_r) * ldA + (k0 + a_c);
        const float4 f0 = *reinterpret_cast<const float4*>(sA);
        const float4 f1 = *reinterpret_cast<const float4*>(sA + 4);
        ra[0]=f0.x; ra[1]=f0.y; ra[2]=f0.z; ra[3]=f0.w;
        ra[4]=f1.x; ra[5]=f1.y; ra[6]=f1.z; ra[7]=f1.w;
        if (MODE == 1) {
            const __bf16* s = Bhb + (long)(col0 + h_n) * ldB + (k0 + h_k);
            rh[0] = reinterpret_cast<const uint4*>(s)[0];
            rh[1] = reinterpret_cast<const uint4*>(s)[1];
        } else {
            #pragma unroll
            for (int j = 0; j < 4; ++j) {
                const float4 q = *reinterpret_cast<const float4*>(
                    Bfb + (long)(k0 + b_k + j) * ldB + (col0 + b_n));
                rb[j][0]=q.x; rb[j][1]=q.y; rb[j][2]=q.z; rb[j][3]=q.w;
            }
        }
        if (k0 + 2 * BK < K)   // speculative prefetch two tiles ahead
            __builtin_prefetch(sA + 2 * BK, 0, 1);
    };

    auto storeTile = [&]() {
        // A: 8 consecutive bf16 -> one b128 store
        uint4 pa;
        pa.x = pk2(ra[0], ra[1]); pa.y = pk2(ra[2], ra[3]);
        pa.z = pk2(ra[4], ra[5]); pa.w = pk2(ra[6], ra[7]);
        *reinterpret_cast<uint4*>(&As[a_r][a_c]) = pa;
        if (MODE == 1) {
            reinterpret_cast<uint4*>(&Bs[h_n][h_k])[0] = rh[0];
            reinterpret_cast<uint4*>(&Bs[h_n][h_k])[1] = rh[1];
        } else {
            // transpose 4k x 4n block: per n, pack 4 k's -> one b64 store
            #pragma unroll
            for (int n = 0; n < 4; ++n) {
                uint2 t;
                t.x = pk2(rb[0][n], rb[1][n]);
                t.y = pk2(rb[2][n], rb[3][n]);
                *reinterpret_cast<uint2*>(&Bs[b_n + n][b_k]) = t;
            }
        }
    };

    loadTile(0);

    for (int k0 = 0; k0 < K; k0 += BK) {
        storeTile();
        __syncthreads();
        if (k0 + BK < K) loadTile(k0 + BK);

        // ---- fragments per ISA 16-bit layouts ----
        Frag16 fa0, fa1, fb0, fb1;
        {
            const int ar0 = warpM * 32 + l16;
            #pragma unroll
            for (int p = 0; p < 8; ++p) {
                const int k = ((p >> 2) << 4) + half * 8 + ((p & 3) << 1);
                fa0.u[p] = *reinterpret_cast<const unsigned*>(&As[ar0][k]);
                fa1.u[p] = *reinterpret_cast<const unsigned*>(&As[ar0 + 16][k]);
            }
            const int kb = half * 8;
            const int n0 = warpN * 32 + l16;
            #pragma unroll
            for (int p = 0; p < 8; ++p) {
                fb0.u[p] = *reinterpret_cast<const unsigned*>(&Bs[n0][kb + ((p >> 2) << 4) + ((p & 3) << 1)]);
                fb1.u[p] = *reinterpret_cast<const unsigned*>(&Bs[n0 + 16][kb + ((p >> 2) << 4) + ((p & 3) << 1)]);
            }
        }

        acc00 = __builtin_amdgcn_wmma_f32_16x16x32_bf16(
                    false, fa0.v, false, fb0.v, (short)0, acc00, false, false);
        acc01 = __builtin_amdgcn_wmma_f32_16x16x32_bf16(
                    false, fa0.v, false, fb1.v, (short)0, acc01, false, false);
        acc10 = __builtin_amdgcn_wmma_f32_16x16x32_bf16(
                    false, fa1.v, false, fb0.v, (short)0, acc10, false, false);
        acc11 = __builtin_amdgcn_wmma_f32_16x16x32_bf16(
                    false, fa1.v, false, fb1.v, (short)0, acc11, false, false);

        __syncthreads();
    }

    // ---- epilogue: C/D layout: VGPR r -> row r + 8*half, col = l16 ----
    const int gcol0 = col0 + warpN * 32 + l16;
    const int gcol1 = gcol0 + 16;
    float b0 = 0.f, b1 = 0.f;
    if constexpr (MODE == 0) { b0 = bias[gcol0]; b1 = bias[gcol1]; }

    #pragma unroll
    for (int r = 0; r < 8; ++r) {
        const int grow0 = row0 + warpM * 32 + r + half * 8;       // frag row block 0
        const int grow1 = grow0 + 16;                              // frag row block 1
        if constexpr (MODE == 0) {
            Ch[(long)grow0 * ldC + gcol0] = (__bf16)(acc00[r] + b0);
            Ch[(long)grow0 * ldC + gcol1] = (__bf16)(acc01[r] + b1);
            Ch[(long)grow1 * ldC + gcol0] = (__bf16)(acc10[r] + b0);
            Ch[(long)grow1 * ldC + gcol1] = (__bf16)(acc11[r] + b1);
        } else {
            float* Cb = Cf + (long)z * cBatch;
            Cb[(long)grow0 * ldC + gcol0] = acc00[r];
            Cb[(long)grow0 * ldC + gcol1] = acc01[r];
            Cb[(long)grow1 * ldC + gcol0] = acc10[r];
            Cb[(long)grow1 * ldC + gcol1] = acc11[r];
        }
    }
}

// In-place softmax along rows of length TKd (= 2048), one block per row.
__global__ __launch_bounds__(256)
void softmax_rows(float* __restrict__ S, int TKd)
{
    __shared__ float red[256];
    float* row = S + (long)blockIdx.x * TKd;
    const int tid = threadIdx.x;

    float v[8];
    float m = -1e30f;
    #pragma unroll
    for (int i = 0; i < 8; ++i) { v[i] = row[tid + i * 256]; m = fmaxf(m, v[i]); }

    red[tid] = m; __syncthreads();
    for (int s = 128; s > 0; s >>= 1) {
        if (tid < s) red[tid] = fmaxf(red[tid], red[tid + s]);
        __syncthreads();
    }
    m = red[0]; __syncthreads();

    float sum = 0.f;
    #pragma unroll
    for (int i = 0; i < 8; ++i) { v[i] = __expf(v[i] - m); sum += v[i]; }

    red[tid] = sum; __syncthreads();
    for (int s = 128; s > 0; s >>= 1) {
        if (tid < s) red[tid] += red[tid + s];
        __syncthreads();
    }
    const float inv = 1.0f / red[0];

    #pragma unroll
    for (int i = 0; i < 8; ++i) row[tid + i * 256] = v[i] * inv;
}

extern "C" void kernel_launch(void* const* d_in, const int* in_sizes, int n_in,
                              void* d_out, int out_size, void* d_ws, size_t ws_size,
                              hipStream_t stream)
{
    (void)in_sizes; (void)n_in; (void)out_size; (void)ws_size;
    constexpr int Bb = 16, TQ = 512, TK = 2048, H = 1024;

    const float* dec  = (const float*)d_in[0];   // [B, TQ, H]
    const float* enc  = (const float*)d_in[1];   // [B, TK, H]
    const float* wa   = (const float*)d_in[2];   // [H, H]
    const float* bias = (const float*)d_in[3];   // [H]

    float* ctx   = (float*)d_out;                        // [B, TQ, H]
    float* align = ctx + (size_t)Bb * TQ * H;            // [B, TQ, TK]
    __bf16* proj = (__bf16*)d_ws;                        // [B*TK, H] bf16 (64 MB)

    // K1: proj = enc @ Wa + bias   (M = B*TK folded, N = H, K = H)
    gemm_bf16_wmma<0><<<dim3(H / BN, (Bb * TK) / BM, 1), 256, 0, stream>>>(
        enc, wa, nullptr, bias, nullptr, proj,
        H, H, /*ldA*/H, /*ldB*/H, /*ldC*/H, 0L, 0L, 0L);

    // K2: score = dec @ proj^T per batch  (M = TQ, N = TK, K = H)
    gemm_bf16_wmma<1><<<dim3(TK / BN, TQ / BM, Bb), 256, 0, stream>>>(
        dec, nullptr, proj, nullptr, align, nullptr,
        TK, H, /*ldA*/H, /*ldB*/H, /*ldC*/TK,
        (long)TQ * H, (long)TK * H, (long)TQ * TK);

    // K3: softmax over TK, in place
    softmax_rows<<<Bb * TQ, 256, 0, stream>>>(align, TK);

    // K4: ctx = align @ enc per batch   (M = TQ, N = H, K = TK)
    gemm_bf16_wmma<2><<<dim3(H / BN, TQ / BM, Bb), 256, 0, stream>>>(
        align, enc, nullptr, nullptr, ctx, nullptr,
        H, TK, /*ldA*/TK, /*ldB*/H, /*ldC*/H,
        (long)TQ * TK, (long)TK * H, (long)TQ * H);
}
